// GATConv_7756710936819
// MI455X (gfx1250) — matrix operand order repaired and checked by reading, not compiled, not run on hardware
//
#include <hip/hip_runtime.h>
#include <stdint.h>

// ---------------------------------------------------------------------------
// GAT dense attention for MI455X (gfx1250, wave32, WMMA f32_16x16x32_f16)
//   N=8192, D_IN=D_HID=D_OUT=256
// Pipeline:
//   1) pack_adj   : A(int32, 256MB) -> bitmask w/ self loops (8MB), read A once
//   2) qkv_gemm x3: Qh,Kh (row-major f16), Vt (transposed f16) via WMMA
//   3) attn_stats : row max m_i / sum Z_i; K tiles staged to LDS with
//                   global_load_async_to_lds_b128 (double buffered, ASYNCcnt)
//   4) attn_out   : O = (exp(S-m)*mask) @ V / Z ; 64 rows/block to cut L2 traffic
// ---------------------------------------------------------------------------

#define NN   8192
#define DD   256

typedef __attribute__((ext_vector_type(16))) _Float16 v16h;
typedef __attribute__((ext_vector_type(8)))  float    v8f;
typedef __attribute__((ext_vector_type(4)))  unsigned int u32x4;

union HFrag {
    v16h     h;
    u32x4    q[2];
    _Float16 e[16];
};

static __device__ __forceinline__ v8f zero8() {
    v8f z = {0.f,0.f,0.f,0.f,0.f,0.f,0.f,0.f};
    return z;
}

// ---------------------------------------------------------------------------
// Kernel 1: pack adjacency into bits, OR in the diagonal (self loops).
// One wave (32 lanes) produces one 32-bit word. Ap[row*256 + w] bit b
// corresponds to column w*32+b.
// ---------------------------------------------------------------------------
__global__ __launch_bounds__(256) void pack_adj(const int* __restrict__ A,
                                                uint32_t* __restrict__ Ap) {
    int tid  = blockIdx.x * blockDim.x + threadIdx.x;
    int lane = tid & 31;
    int word = tid >> 5;              // global word index, < N*256
    int row  = word >> 8;             // 256 words per row
    int w    = word & 255;
    int col  = (w << 5) + lane;
    int a    = A[(size_t)row * NN + col];
    bool bit = (a != 0) || (col == row);
    unsigned long long m = __ballot(bit);   // wave32: low 32 bits valid
    if (lane == 0) Ap[word] = (uint32_t)m;
}

// ---------------------------------------------------------------------------
// Kernel 2: Out[r*rs + c*cs] = f16( (X @ W)[r,c] ), via WMMA.
// W (f32 256x256) is transposed into LDS as f16 with padded row stride
// (264 halves = 528B) so B-fragment ds_load_b128s are bank-conflict free.
// Workgroup: 8 waves x 16 rows = 128 rows. Grid: 64 blocks.
// ---------------------------------------------------------------------------
__global__ __launch_bounds__(256) void qkv_gemm(const float* __restrict__ X,
                                                const float* __restrict__ W,
                                                _Float16* __restrict__ Out,
                                                long rstride, long cstride) {
    extern __shared__ char smem[];
    _Float16* Wt = (_Float16*)smem;         // Wt[c*264 + k] = W[k,c]
    const int WT_STRIDE = 264;

    int t = threadIdx.x;
    for (int idx = t; idx < DD * DD; idx += 256) {
        int k = idx >> 8, c = idx & 255;
        Wt[c * WT_STRIDE + k] = (_Float16)W[idx];
    }
    __syncthreads();

    int wave = t >> 5, lane = t & 31;
    int half = lane >> 4, ln = lane & 15;
    int row0 = blockIdx.x * 128 + wave * 16;

    // A fragments: X rows row0..row0+15, f32 -> f16.
    HFrag a[8];
    for (int kc = 0; kc < 8; ++kc) {
        const float* xp = X + (size_t)(row0 + ln) * DD + kc * 32 + half * 8;
        float4 f0 = *(const float4*)(xp);
        float4 f1 = *(const float4*)(xp + 4);
        float4 f2 = *(const float4*)(xp + 16);
        float4 f3 = *(const float4*)(xp + 20);
        a[kc].e[0]=(_Float16)f0.x;  a[kc].e[1]=(_Float16)f0.y;
        a[kc].e[2]=(_Float16)f0.z;  a[kc].e[3]=(_Float16)f0.w;
        a[kc].e[4]=(_Float16)f1.x;  a[kc].e[5]=(_Float16)f1.y;
        a[kc].e[6]=(_Float16)f1.z;  a[kc].e[7]=(_Float16)f1.w;
        a[kc].e[8]=(_Float16)f2.x;  a[kc].e[9]=(_Float16)f2.y;
        a[kc].e[10]=(_Float16)f2.z; a[kc].e[11]=(_Float16)f2.w;
        a[kc].e[12]=(_Float16)f3.x; a[kc].e[13]=(_Float16)f3.y;
        a[kc].e[14]=(_Float16)f3.z; a[kc].e[15]=(_Float16)f3.w;
    }

    for (int nt = 0; nt < 16; ++nt) {
        int c0 = nt * 16;
        v8f acc = zero8();
        for (int kc = 0; kc < 8; ++kc) {
            HFrag b;
            const _Float16* wp = Wt + (size_t)(c0 + ln) * WT_STRIDE + kc * 32 + half * 8;
            b.q[0] = *(const u32x4*)(wp);
            b.q[1] = *(const u32x4*)(wp + 16);
            acc = __builtin_amdgcn_wmma_f32_16x16x32_f16(false, a[kc].h, false, b.h,
                                                         (short)0, acc, false, false);
        }
        for (int v = 0; v < 8; ++v) {
            long r = row0 + v + 8 * half;
            Out[r * rstride + (long)(c0 + ln) * cstride] = (_Float16)acc[v];
        }
    }
}

// ---------------------------------------------------------------------------
// Async staging: copy K[jb..jb+64) x 256 halves into LDS (row stride 528B)
// with global_load_async_to_lds_b128. 256 threads x 8 iters x 16B = 32KB.
// Each wave issues 8 async ops (ASYNCcnt += 8).
// ---------------------------------------------------------------------------
__device__ __forceinline__ void stage_k_async(const _Float16* __restrict__ Kh,
                                              _Float16* kbuf, int jb, int t) {
#pragma unroll
    for (int it = 0; it < 8; ++it) {
        int i = t + it * 256;
        int row = i >> 5, chunk = i & 31;           // 64 rows x 32 chunks(16B)
        uint32_t loff = (uint32_t)(uintptr_t)kbuf + (uint32_t)(row * 528 + chunk * 16);
        uint64_t ga = (uint64_t)(uintptr_t)(Kh + (size_t)(jb + row) * DD + chunk * 8);
        asm volatile("global_load_async_to_lds_b128 %0, %1, off"
                     :: "v"(loff), "v"(ga) : "memory");
    }
}

// ---------------------------------------------------------------------------
// Kernel 3: per-row softmax stats over allowed columns.
// Workgroup: 8 waves x 16 rows = 128 rows. Grid: 64 blocks.
// K tiles double-buffered in LDS via async loads; each wave streams all
// 8192 columns in 16-wide S tiles (8 WMMAs each), keeps online (max,sum),
// then merges across the 16 lanes of each half.
// ---------------------------------------------------------------------------
__global__ __launch_bounds__(256) void attn_stats(const _Float16* __restrict__ Qh,
                                                  const _Float16* __restrict__ Kh,
                                                  const uint32_t* __restrict__ Ap,
                                                  float* __restrict__ Mrow,
                                                  float* __restrict__ Zrow) {
    extern __shared__ _Float16 Kdyn[];               // 2 x 64 x 264 halves = 66KB
    const int KSTRIDE = 264;                         // halves (528B, 33x16B)
    _Float16* kb0 = Kdyn;
    _Float16* kb1 = Kdyn + 64 * KSTRIDE;

    int t = threadIdx.x;
    int wave = t >> 5, lane = t & 31;
    int half = lane >> 4, ln = lane & 15;
    int row0 = blockIdx.x * 128 + wave * 16;

    // Resident Q fragments (A operand): 16 rows x 256 feat = 64 VGPRs
    HFrag qa[8];
    for (int kc = 0; kc < 8; ++kc) {
        const _Float16* qp = Qh + (size_t)(row0 + ln) * DD + kc * 32 + half * 8;
        qa[kc].q[0] = *(const u32x4*)(qp);
        qa[kc].q[1] = *(const u32x4*)(qp + 16);
    }

    float m[8], z[8];
    for (int v = 0; v < 8; ++v) { m[v] = -3.0e38f; z[v] = 0.0f; }

    stage_k_async(Kh, kb0, 0, t);                    // prologue: buffer 0

    for (int jb = 0; jb < NN; jb += 64) {
        int cur = (jb >> 6) & 1;
        _Float16* kcur = cur ? kb1 : kb0;
        if (jb + 64 < NN) {
            stage_k_async(Kh, cur ? kb0 : kb1, jb + 64, t);
            asm volatile("s_wait_asynccnt 0x8" ::: "memory");   // current buf done
        } else {
            asm volatile("s_wait_asynccnt 0x0" ::: "memory");
        }
        __syncthreads();                             // all waves' asyncs visible

        for (int jt = 0; jt < 4; ++jt) {
            v8f s = zero8();
            for (int kc = 0; kc < 8; ++kc) {
                // B = K^T fragment from LDS: lane holds col n=ln -> K row jt*16+ln
                HFrag b;
                const _Float16* kp = kcur + (size_t)(jt * 16 + ln) * KSTRIDE
                                          + kc * 32 + half * 8;
                b.q[0] = *(const u32x4*)(kp);
                b.q[1] = *(const u32x4*)(kp + 16);
                s = __builtin_amdgcn_wmma_f32_16x16x32_f16(false, qa[kc].h, false, b.h,
                                                           (short)0, s, false, false);
            }
            int col = jb + jt * 16 + ln;
            int w = col >> 5, bitn = col & 31;
            for (int v = 0; v < 8; ++v) {
                int r = row0 + v + 8 * half;
                uint32_t word = Ap[(size_t)r * 256 + w];
                bool ok = (word >> bitn) & 1u;
                float sv = ok ? s[v] : -__builtin_inff();
                float mn = fmaxf(m[v], sv);
                z[v] = z[v] * __expf(m[v] - mn) + __expf(sv - mn);
                m[v] = mn;
            }
        }
        __syncthreads();                             // done reading before refill
    }

    // Merge across the 16 lanes of each half (xor 1,2,4,8 stays inside a half)
    for (int off = 1; off < 16; off <<= 1) {
        for (int v = 0; v < 8; ++v) {
            float mo = __shfl_xor(m[v], off, 32);
            float zo = __shfl_xor(z[v], off, 32);
            float mn = fmaxf(m[v], mo);
            z[v] = z[v] * __expf(m[v] - mn) + zo * __expf(mo - mn);
            m[v] = mn;
        }
    }
    if (ln == 0) {
        for (int v = 0; v < 8; ++v) {
            int r = row0 + v + 8 * half;
            Mrow[r] = m[v];
            Zrow[r] = z[v];
        }
    }
}

// ---------------------------------------------------------------------------
// Kernel 4: O = (exp(S - m) * mask) @ V / Z.
// Workgroup: 64 rows, 8 waves = 4 row-groups (rg) x 2 col-groups (cg).
// Per 32-col step: wave (rg,cg) computes the S tile (rows rg, cols jb+16cg),
// exp+mask -> f16 P tile in LDS; barrier; wave consumes P[rg][0..1] as one
// 16x32 A fragment for 8 P@V WMMAs into its 16x128 accumulator.
// Grid: 128 blocks (cuts Kh/Vt L2 traffic ~4x vs 16-row blocks).
// ---------------------------------------------------------------------------
__global__ __launch_bounds__(256) void attn_out(const _Float16* __restrict__ Qh,
                                                const _Float16* __restrict__ Kh,
                                                const _Float16* __restrict__ Vt,
                                                const uint32_t* __restrict__ Ap,
                                                const float* __restrict__ Mrow,
                                                const float* __restrict__ Zrow,
                                                float* __restrict__ Out) {
    const int PSTRIDE = 24;                          // 48B rows: conflict-free b128
    __shared__ _Float16 Plds[4][2][16 * PSTRIDE];    // 6 KB

    int t = threadIdx.x;
    int wave = t >> 5, lane = t & 31;
    int half = lane >> 4, ln = lane & 15;
    int rg = wave >> 1, cg = wave & 1;
    int row0 = blockIdx.x * 64 + rg * 16;
    int cbase = cg * 128;

    HFrag qa[8];
    for (int kc = 0; kc < 8; ++kc) {
        const _Float16* qp = Qh + (size_t)(row0 + ln) * DD + kc * 32 + half * 8;
        qa[kc].q[0] = *(const u32x4*)(qp);
        qa[kc].q[1] = *(const u32x4*)(qp + 16);
    }
    float mloc[8];
    for (int v = 0; v < 8; ++v) mloc[v] = Mrow[row0 + v + 8 * half];

    v8f acc[8];
    for (int nt = 0; nt < 8; ++nt) acc[nt] = zero8();

    for (int jb = 0; jb < NN; jb += 32) {
        int j = jb + cg * 16;
        // ---- S tile for cols j..j+15 ----
        v8f s = zero8();
        for (int kc = 0; kc < 8; ++kc) {
            HFrag b;
            const _Float16* kp = Kh + (size_t)(j + ln) * DD + kc * 32 + half * 8;
            b.q[0] = *(const u32x4*)(kp);
            b.q[1] = *(const u32x4*)(kp + 16);
            s = __builtin_amdgcn_wmma_f32_16x16x32_f16(false, qa[kc].h, false, b.h,
                                                       (short)0, s, false, false);
        }
        // prefetch next K tile row for this lane (global_prefetch_b8)
        if (jb + 32 < NN)
            __builtin_prefetch(Kh + (size_t)(j + 32 + ln) * DD, 0, 1);

        // ---- mask + exp -> LDS (f16), C layout: (row v+8*half, col ln) ----
        int col = j + ln;
        int w = col >> 5, bitn = col & 31;
        for (int v = 0; v < 8; ++v) {
            int r = row0 + v + 8 * half;
            uint32_t word = Ap[(size_t)r * 256 + w];
            bool ok = (word >> bitn) & 1u;
            float p = ok ? __expf(s[v] - mloc[v]) : 0.0f;
            Plds[rg][cg][(v + 8 * half) * PSTRIDE + ln] = (_Float16)p;
        }
        __syncthreads();

        // ---- P @ V: one 16x32 A fragment, 8 WMMAs over 128 output cols ----
        HFrag pa;
        pa.q[0] = *(const u32x4*)(&Plds[rg][0][ln * PSTRIDE + half * 8]);
        pa.q[1] = *(const u32x4*)(&Plds[rg][1][ln * PSTRIDE + half * 8]);
        for (int nt = 0; nt < 8; ++nt) {
            HFrag vb;
            const _Float16* vp = Vt + (size_t)(cbase + nt * 16 + ln) * NN + jb + half * 8;
            vb.q[0] = *(const u32x4*)(vp);
            vb.q[1] = *(const u32x4*)(vp + 16);
            acc[nt] = __builtin_amdgcn_wmma_f32_16x16x32_f16(false, pa.h, false, vb.h,
                                                             (short)0, acc[nt], false, false);
        }
        __syncthreads();
    }

    // ---- divide by Z and store ----
    for (int v = 0; v < 8; ++v) {
        int r = row0 + v + 8 * half;
        float zinv = 1.0f / Zrow[r];
        for (int nt = 0; nt < 8; ++nt)
            Out[(size_t)r * DD + cbase + nt * 16 + ln] = acc[nt][v] * zinv;
    }
}

// ---------------------------------------------------------------------------
// Launch
// ---------------------------------------------------------------------------
extern "C" void kernel_launch(void* const* d_in, const int* in_sizes, int n_in,
                              void* d_out, int out_size, void* d_ws, size_t ws_size,
                              hipStream_t stream) {
    const float* X  = (const float*)d_in[0];
    const int*   A  = (const int*)  d_in[1];
    const float* Wq = (const float*)d_in[2];
    const float* Wk = (const float*)d_in[3];
    const float* Wv = (const float*)d_in[4];
    float* Out = (float*)d_out;

    char* ws = (char*)d_ws;
    _Float16* Qh = (_Float16*)(ws);                          // 4 MB
    _Float16* Kh = (_Float16*)(ws + (4u  << 20));            // 4 MB
    _Float16* Vt = (_Float16*)(ws + (8u  << 20));            // 4 MB (transposed)
    uint32_t* Ap = (uint32_t*)(ws + (12u << 20));            // 8 MB
    float* Mrow  = (float*)   (ws + (20u << 20));            // 32 KB
    float* Zrow  = (float*)   (ws + (20u << 20) + 32768);    // 32 KB

    // 1) pack adjacency (+ self loops): one bit word per wave32
    pack_adj<<<(NN / 1) * (NN / 256), 256, 0, stream>>>(A, Ap);

    // 2) QKV GEMMs (V stored transposed)
    {
        size_t lds = (size_t)256 * 264 * sizeof(_Float16);   // 132 KB
        qkv_gemm<<<64, 256, lds, stream>>>(X, Wq, Qh, 256, 1);
        qkv_gemm<<<64, 256, lds, stream>>>(X, Wk, Kh, 256, 1);
        qkv_gemm<<<64, 256, lds, stream>>>(X, Wv, Vt, 1, NN);
    }

    // 3) softmax stats (async double-buffered K staging: 2 x 64 x 528B)
    {
        size_t lds = (size_t)2 * 64 * 264 * sizeof(_Float16);   // 67584 B
        attn_stats<<<64, 256, lds, stream>>>(Qh, Kh, Ap, Mrow, Zrow);
    }

    // 4) masked attention output
    attn_out<<<128, 256, 0, stream>>>(Qh, Kh, Vt, Ap, Mrow, Zrow, Out);
}
